// Model_19550691131427
// MI455X (gfx1250) — matrix operand order repaired and checked by reading, not compiled, not run on hardware
//
#include <hip/hip_runtime.h>
#include <math.h>

typedef __attribute__((ext_vector_type(16))) _Float16 v16h;
typedef __attribute__((ext_vector_type(8)))  _Float16 v8h;
typedef __attribute__((ext_vector_type(8)))  float    v8f;

#define L_DIM 2
#define N_IMG 8          // N1 == N2 == 8
#define P_DIM 1024
#define D_DIM 768
#define ROWS_PER_LAYER (N_IMG * P_DIM)   // 8192

// Build a 16-half WMMA fragment from two 8-half (16B) chunks.
__device__ __forceinline__ v16h load_frag(const _Float16* p0, const _Float16* p1) {
    v8h lo = *(const v8h*)p0;
    v8h hi = *(const v8h*)p1;
    v16h r;
    #pragma unroll
    for (int i = 0; i < 8; ++i) { r[i] = lo[i]; r[8 + i] = hi[i]; }
    return r;
}

// ---------------------------------------------------------------------------
// 1) Row L2-normalize (f32) -> quantize to f16, and emit sum-of-squares of the
//    quantized row (so the distance identity is self-consistent in f16).
//    One wave32 per row; 8 rows per 256-thread block.
// ---------------------------------------------------------------------------
__global__ __launch_bounds__(256) void normalize_kernel(
    const float* __restrict__ src,   // [nrows][768] f32
    _Float16*    __restrict__ dsth,  // [nrows][768] f16
    float*       __restrict__ sq,    // [nrows] f32
    int nrows)
{
    const int lane = threadIdx.x & 31;
    const int wv   = threadIdx.x >> 5;
    const int row  = blockIdx.x * 8 + wv;
    if (row >= nrows) return;

    const float* s = src + (size_t)row * D_DIM;
    float acc = 0.f;
    #pragma unroll
    for (int i = lane; i < D_DIM; i += 32) { float v = s[i]; acc += v * v; }
    #pragma unroll
    for (int off = 16; off; off >>= 1) acc += __shfl_xor(acc, off, 32);
    const float rinv = 1.0f / sqrtf(acc);

    _Float16* d = dsth + (size_t)row * D_DIM;
    float acc2 = 0.f;
    #pragma unroll
    for (int i = lane; i < D_DIM; i += 32) {
        _Float16 h = (_Float16)(s[i] * rinv);
        d[i] = h;
        float hv = (float)h;
        acc2 += hv * hv;
    }
    #pragma unroll
    for (int off = 16; off; off >>= 1) acc2 += __shfl_xor(acc2, off, 32);
    if (lane == 0) sq[row] = acc2;
}

// ---------------------------------------------------------------------------
// 2) Register-blocked WMMA GEMM with fused per-(row, n2-block) min reduction.
//    Each wave: 32 rows x 64 cols per block iteration (2 row-tiles x 4
//    col-tiles -> 8 WMMAs per K-step from 2 A + 4 B fragments = 1.5 loads/WMMA).
//    sp[l][n1][n2][p] = mask[n1][p] * sqrt(max(a2 + min_q(b2[q]-2*dot), 1e-12))/2
// ---------------------------------------------------------------------------
__global__ __launch_bounds__(128) void gemm_min_kernel(
    const _Float16* __restrict__ fA,   // [L][8192][768]
    const _Float16* __restrict__ fB,   // [L][8192][768]
    const float*    __restrict__ a2,   // [L][8192]
    const float*    __restrict__ b2,   // [L][8192]
    const float*    __restrict__ mask, // [8][1024]
    float*          __restrict__ sp)   // [L][8][8][1024]
{
    const int lane  = threadIdx.x & 31;
    const int wave  = threadIdx.x >> 5;
    const int l     = blockIdx.z;
    const int n2    = blockIdx.y;
    const int rowbase = blockIdx.x * 128 + wave * 32;   // two 16-row tiles
    const int half  = lane >> 4;                         // 0: lanes 0-15, 1: lanes 16-31
    const int lid   = lane & 15;

    const _Float16* Arow0 = fA + ((size_t)l * ROWS_PER_LAYER + rowbase + lid) * D_DIM;
    const _Float16* Arow1 = Arow0 + (size_t)16 * D_DIM;
    const _Float16* Bbase = fB + ((size_t)l * ROWS_PER_LAYER + n2 * P_DIM) * D_DIM;
    const float*    b2l   = b2 + (size_t)l * ROWS_PER_LAYER + n2 * P_DIM;

    float rowmin[2][8];
    #pragma unroll
    for (int m = 0; m < 2; ++m)
        #pragma unroll
        for (int r = 0; r < 8; ++r) rowmin[m][r] = 3.0e38f;

    for (int cg = 0; cg < P_DIM / 64; ++cg) {            // 16 col groups of 64
        const int colbase = cg * 64;
        const _Float16* Brow[4];
        #pragma unroll
        for (int t = 0; t < 4; ++t) {
            Brow[t] = Bbase + (size_t)(colbase + t * 16 + lid) * D_DIM + half * 8;
            if (cg + 1 < P_DIM / 64)                      // warm L0 for next col group
                __builtin_prefetch(Brow[t] + (size_t)64 * D_DIM, 0, 3);
        }

        v8f c[2][4];
        #pragma unroll
        for (int m = 0; m < 2; ++m)
            #pragma unroll
            for (int t = 0; t < 4; ++t) c[m][t] = (v8f){};

        for (int kk = 0; kk < D_DIM / 32; ++kk) {         // 24 K-steps
            const int kb = kk * 32;
            // A fragments: lanes 0-15 K base 0, lanes 16-31 K base 8; chunks +0,+16.
            const _Float16* ap0 = Arow0 + kb + half * 8;
            const _Float16* ap1 = Arow1 + kb + half * 8;
            v16h a0 = load_frag(ap0, ap0 + 16);
            v16h a1 = load_frag(ap1, ap1 + 16);
            #pragma unroll
            for (int t = 0; t < 4; ++t) {
                // B fragment: column = lid, K base half*8; chunks +0,+16 (matches A).
                const _Float16* bp = Brow[t] + kb;
                v16h b = load_frag(bp, bp + 16);
                c[0][t] = __builtin_amdgcn_wmma_f32_16x16x32_f16(
                              false, a0, false, b, (short)0, c[0][t], false, false);
                c[1][t] = __builtin_amdgcn_wmma_f32_16x16x32_f16(
                              false, a1, false, b, (short)0, c[1][t], false, false);
            }
        }

        // Epilogue: acc VGPR r of tile (m,t) holds (M = m*16 + r + half*8, N = t*16 + lid).
        #pragma unroll
        for (int t = 0; t < 4; ++t) {
            const float bb = b2l[colbase + t * 16 + lid];
            #pragma unroll
            for (int r = 0; r < 8; ++r) {
                rowmin[0][r] = fminf(rowmin[0][r], bb - 2.0f * c[0][t][r]);
                rowmin[1][r] = fminf(rowmin[1][r], bb - 2.0f * c[1][t][r]);
            }
        }
    }

    // Min across the 16 column-lanes within each half-wave (butterfly).
    #pragma unroll
    for (int m = 0; m < 2; ++m)
        #pragma unroll
        for (int r = 0; r < 8; ++r) {
            float v = rowmin[m][r];
            #pragma unroll
            for (int off = 1; off < 16; off <<= 1) v = fminf(v, __shfl_xor(v, off, 32));
            rowmin[m][r] = v;
        }

    if (lid == 0) {
        const float* a2l = a2 + (size_t)l * ROWS_PER_LAYER;
        #pragma unroll
        for (int m = 0; m < 2; ++m)
            #pragma unroll
            for (int r = 0; r < 8; ++r) {
                const int row = rowbase + m * 16 + r + half * 8;
                const int p   = row & (P_DIM - 1);
                const int n1  = row >> 10;
                float d2 = a2l[row] + rowmin[m][r];
                float d  = sqrtf(fmaxf(d2, 1e-12f)) * 0.5f;
                sp[((((size_t)l * N_IMG) + n1) * N_IMG + n2) * P_DIM + p] =
                    d * mask[n1 * P_DIM + p];
            }
    }
}

// ---------------------------------------------------------------------------
// 3) scores[n1] = mean_{l,n2} max_p sp[l][n1][n2][p]
// ---------------------------------------------------------------------------
__global__ __launch_bounds__(256) void scores_kernel(
    const float* __restrict__ sp, float* __restrict__ out)
{
    __shared__ float red[8];
    const int n1 = blockIdx.x;
    const int tid = threadIdx.x;
    const int lane = tid & 31, wv = tid >> 5;
    float acc = 0.f;
    for (int c = 0; c < L_DIM * N_IMG; ++c) {
        const int l = c >> 3, nn2 = c & 7;
        const float* p = sp + ((((size_t)l * N_IMG) + n1) * N_IMG + nn2) * P_DIM;
        float m = -3.0e38f;
        for (int i = tid; i < P_DIM; i += 256) m = fmaxf(m, p[i]);
        #pragma unroll
        for (int off = 16; off; off >>= 1) m = fmaxf(m, __shfl_xor(m, off, 32));
        if (lane == 0) red[wv] = m;
        __syncthreads();
        if (tid == 0) {
            float mm = red[0];
            #pragma unroll
            for (int w = 1; w < 8; ++w) mm = fmaxf(mm, red[w]);
            acc += mm;
        }
        __syncthreads();
    }
    if (tid == 0) out[n1] = acc * (1.0f / (L_DIM * N_IMG));
}

// ---------------------------------------------------------------------------
// 4) spatch[n1][p] = mean_{l,n2} sp[l][n1][n2][p]
// ---------------------------------------------------------------------------
__global__ __launch_bounds__(256) void spatch_kernel(
    const float* __restrict__ sp, float* __restrict__ spatch)
{
    const int idx = blockIdx.x * 256 + threadIdx.x;      // 8 * 1024
    if (idx >= N_IMG * P_DIM) return;
    const int n1 = idx >> 10, p = idx & (P_DIM - 1);
    float s = 0.f;
    #pragma unroll
    for (int l = 0; l < L_DIM; ++l)
        #pragma unroll
        for (int nn2 = 0; nn2 < N_IMG; ++nn2)
            s += sp[((((size_t)l * N_IMG) + n1) * N_IMG + nn2) * P_DIM + p];
    spatch[idx] = s * (1.0f / (L_DIM * N_IMG));
}

// ---------------------------------------------------------------------------
// 5) Bilinear 32x32 -> 512x512, half-pixel centers, edge clamp.
// ---------------------------------------------------------------------------
__global__ __launch_bounds__(256) void upsample_kernel(
    const float* __restrict__ spatch, float* __restrict__ outpix)
{
    const int idx = blockIdx.x * 256 + threadIdx.x;      // 8 * 512 * 512
    if (idx >= N_IMG * 512 * 512) return;
    const int x  = idx & 511;
    const int y  = (idx >> 9) & 511;
    const int n1 = idx >> 18;
    const float fy = (y + 0.5f) * (1.0f / 16.0f) - 0.5f;
    const float fx = (x + 0.5f) * (1.0f / 16.0f) - 0.5f;
    const int y0 = (int)floorf(fy);
    const int x0 = (int)floorf(fx);
    const float wy = fy - (float)y0;
    const float wx = fx - (float)x0;
    const int y0c = min(max(y0, 0), 31),     y1c = min(max(y0 + 1, 0), 31);
    const int x0c = min(max(x0, 0), 31),     x1c = min(max(x0 + 1, 0), 31);
    const float* S = spatch + n1 * P_DIM;
    const float v =
        (1.f - wy) * ((1.f - wx) * S[y0c * 32 + x0c] + wx * S[y0c * 32 + x1c]) +
        wy         * ((1.f - wx) * S[y1c * 32 + x0c] + wx * S[y1c * 32 + x1c]);
    outpix[idx] = v;
}

// ---------------------------------------------------------------------------
extern "C" void kernel_launch(void* const* d_in, const int* in_sizes, int n_in,
                              void* d_out, int out_size, void* d_ws, size_t ws_size,
                              hipStream_t stream) {
    const float* feats  = (const float*)d_in[0];  // [2,8,1024,768] f32
    const float* nfeats = (const float*)d_in[1];  // [2,8,1024,768] f32
    const float* mask   = (const float*)d_in[2];  // [8,1024] f32
    // d_in[3..5] = img_h(512), img_w(512), patch_size(16): compile-time constants here.

    char* ws = (char*)d_ws;
    const size_t halfBytes = (size_t)L_DIM * ROWS_PER_LAYER * D_DIM * sizeof(_Float16); // 25,165,824
    _Float16* fA     = (_Float16*)(ws);
    _Float16* fB     = (_Float16*)(ws + halfBytes);
    float*    a2     = (float*)(ws + 2 * halfBytes);
    float*    b2     = (float*)(ws + 2 * halfBytes + 65536);
    float*    sp     = (float*)(ws + 2 * halfBytes + 2 * 65536);
    float*    spatch = (float*)(ws + 2 * halfBytes + 2 * 65536 + 524288);

    const int nrows = L_DIM * ROWS_PER_LAYER;            // 16384
    normalize_kernel<<<nrows / 8, 256, 0, stream>>>(feats,  fA, a2, nrows);
    normalize_kernel<<<nrows / 8, 256, 0, stream>>>(nfeats, fB, b2, nrows);

    dim3 g(ROWS_PER_LAYER / 128, N_IMG, L_DIM);          // 64 x 8 x 2
    gemm_min_kernel<<<g, 128, 0, stream>>>(fA, fB, a2, b2, mask, sp);

    float* out = (float*)d_out;
    scores_kernel<<<N_IMG, 256, 0, stream>>>(sp, out);
    spatch_kernel<<<(N_IMG * P_DIM + 255) / 256, 256, 0, stream>>>(sp, spatch);
    upsample_kernel<<<(N_IMG * 512 * 512 + 255) / 256, 256, 0, stream>>>(spatch, out + N_IMG);
}
